// EnhancedLSTMCell_58866821759063
// MI455X (gfx1250) — matrix operand order repaired and checked by reading, not compiled, not run on hardware
//
#include <hip/hip_runtime.h>
#include <hip/hip_bf16.h>

// ---------------------------------------------------------------------------
// EnhancedLSTMCell for MI455X (gfx1250, wave32, WMMA)
//   B=8192, IN=1024, H=1024, K=IN+H=2048, 4H=4096
//   gates = [x|h] @ W + b  -> fused gate nonlinearity + cell update
//   then LayerNorm(c_t) and h_t = o * tanh(c_t)
// GEMM path: BF16 operands, v_wmma_f32_16x16x32_bf16 accumulate in F32.
// ---------------------------------------------------------------------------

#define LSTM_B   8192
#define LSTM_K   2048   // IN + H
#define LSTM_H   1024
#define LSTM_4H  4096

typedef __attribute__((ext_vector_type(16))) __bf16 v16bf;
typedef __attribute__((ext_vector_type(8)))  __bf16 v8bf;
typedef __attribute__((ext_vector_type(8)))  float  v8f;

union BF16x16 { v16bf v; v8bf h[2]; };

__device__ __forceinline__ float sigmoid_f(float x) {
    return 1.0f / (1.0f + __expf(-x));
}

// WGP-scope prefetch of (p + 256B) = the NEXT 256-byte group of this stream.
// Scope field omitted => SCOPE 0 = WGP (fetch into every cache level
// on a miss); __builtin_prefetch only reaches SE/SYS scope which skips the WGP
// cache (ISA 7.3 prefetch TH table).  Takes the live loop pointer directly so
// no address VALU is materialized; no memory clobber so it schedules freely.
__device__ __forceinline__ void prefetch_wgp_nextgroup(const void* p) {
    asm volatile("global_prefetch_b8 %0, off offset:256" :: "v"(p));
}

// ---------------------------------------------------------------------------
// Kernel 1: pack [x | h_prev] rows into BF16 A[8192 x 2048]
// ---------------------------------------------------------------------------
__global__ __launch_bounds__(256) void lstm_convA(
    const float* __restrict__ x, const float* __restrict__ hprev,
    __hip_bfloat16* __restrict__ A)
{
    const int idx = blockIdx.x * 256 + threadIdx.x;     // over 8192*2048
    const int r = idx >> 11;          // row
    const int c = idx & 2047;         // col in [0, 2048)
    float v = (c < LSTM_H) ? x[(size_t)r * LSTM_H + c]
                           : hprev[(size_t)r * LSTM_H + (c - LSTM_H)];
    A[idx] = __float2bfloat16(v);
}

// ---------------------------------------------------------------------------
// Kernel 2: LDS-tiled transpose W[2048 x 4096] f32 -> Wt[4096 x 2048] bf16
// (so that for output column n, the K dimension is contiguous)
// ---------------------------------------------------------------------------
__global__ __launch_bounds__(256) void lstm_transposeW(
    const float* __restrict__ W, __hip_bfloat16* __restrict__ Wt)
{
    __shared__ float tile[32][33];
    const int n0 = blockIdx.x * 32;   // over 4096
    const int k0 = blockIdx.y * 32;   // over 2048
    const int tx = threadIdx.x;       // 0..31
    const int ty = threadIdx.y;       // 0..7
#pragma unroll
    for (int j = 0; j < 4; ++j) {
        int k = k0 + ty + j * 8;
        tile[ty + j * 8][tx] = W[(size_t)k * LSTM_4H + n0 + tx];
    }
    __syncthreads();
#pragma unroll
    for (int j = 0; j < 4; ++j) {
        int n = n0 + ty + j * 8;
        Wt[(size_t)n * LSTM_K + k0 + tx] = __float2bfloat16(tile[tx][ty + j * 8]);
    }
}

// ---------------------------------------------------------------------------
// Kernel 3: WMMA GEMM + fused gate epilogue.
//   Block = 256 threads = 8 waves. Macro-tile: 128 rows x 32 h-cols.
//   wave w: m_pair = w&3  -> TWO 16-row slices (rows p*16.. and 64+p*16..),
//           n_half = w>>2 -> 16 h-cols,
//   computes the SAME h-slice for all 4 gate blocks (i,f,g,o), so each B
//   fragment feeds 2 WMMAs and the cell update fuses in-register.
//   Fragment layouts (wave32):
//     A 16x32 bf16: lane: M=lane&15, 16 contiguous K at 16*(lane>>4).
//     B 32x16 bf16: lane: N=lane&15, 16 contiguous K at 16*(lane>>4).
//     C/D 16x16 f32: elem j: M = j + 8*(lane>>4), N = lane&15.
//   K-loop: outer stride = 128 elems (4 WMMA steps, 256 B per stream) with a
//   fully unrolled 4-step inner body; prefetches of the NEXT group issue once
//   per outer iteration on the live pointers (no address VALU, no WMMA->VALU
//   hazard nops in the unrolled body).
// ---------------------------------------------------------------------------
__global__ __launch_bounds__(256) void lstm_gemm_gates(
    const __hip_bfloat16* __restrict__ A,    // [8192 x 2048]
    const __hip_bfloat16* __restrict__ Wt,   // [4096 x 2048]
    const float* __restrict__ bias,          // [4096]
    const float* __restrict__ c_prev,        // [8192 x 1024]
    const float* __restrict__ fmask,         // [8192]
    float* __restrict__ c_pre,               // [8192 x 1024]
    float* __restrict__ o_gate)              // [8192 x 1024]
{
    const int lane = threadIdx.x & 31;
    const int wave = threadIdx.x >> 5;
    const int lrow = lane & 15;
    const int lhi  = lane >> 4;

    const int mA = blockIdx.y * 128 + (wave & 3) * 16;  // first 16-row slice
    const int mB = mA + 64;                             // second 16-row slice
    const int h0 = blockIdx.x * 32 + (wave >> 2) * 16;  // h base of tile

    // Per-lane A pointers: row m+lrow, K chunks at lhi*16 (+0 and +8 in-loop)
    const __hip_bfloat16* aptrA = A + (size_t)(mA + lrow) * LSTM_K + lhi * 16;
    const __hip_bfloat16* aptrB = A + (size_t)(mB + lrow) * LSTM_K + lhi * 16;
    // Per-lane B pointers, one per gate block
    const __hip_bfloat16* bptr0 = Wt + (size_t)(0 * LSTM_H + h0 + lrow) * LSTM_K + lhi * 16;
    const __hip_bfloat16* bptr1 = Wt + (size_t)(1 * LSTM_H + h0 + lrow) * LSTM_K + lhi * 16;
    const __hip_bfloat16* bptr2 = Wt + (size_t)(2 * LSTM_H + h0 + lrow) * LSTM_K + lhi * 16;
    const __hip_bfloat16* bptr3 = Wt + (size_t)(3 * LSTM_H + h0 + lrow) * LSTM_K + lhi * 16;

    v8f accA0 = {}, accA1 = {}, accA2 = {}, accA3 = {};
    v8f accB0 = {}, accB1 = {}, accB2 = {}, accB3 = {};

#pragma unroll 1
    for (int kt = 0; kt < LSTM_K; kt += 128) {
        // Prefetch the next 256B group of every stream into the WGP cache.
        prefetch_wgp_nextgroup(aptrA);
        prefetch_wgp_nextgroup(aptrB);
        prefetch_wgp_nextgroup(bptr0);
        prefetch_wgp_nextgroup(bptr1);
        prefetch_wgp_nextgroup(bptr2);
        prefetch_wgp_nextgroup(bptr3);

#pragma unroll
        for (int u = 0; u < 4; ++u) {
            const int ke = u * 32;    // element offset within the group

            BF16x16 a0, a1;
            a0.h[0] = *(const v8bf*)(aptrA + ke);
            a0.h[1] = *(const v8bf*)(aptrA + ke + 8);
            a1.h[0] = *(const v8bf*)(aptrB + ke);
            a1.h[1] = *(const v8bf*)(aptrB + ke + 8);

            BF16x16 b0, b1, b2, b3;
            b0.h[0] = *(const v8bf*)(bptr0 + ke); b0.h[1] = *(const v8bf*)(bptr0 + ke + 8);
            b1.h[0] = *(const v8bf*)(bptr1 + ke); b1.h[1] = *(const v8bf*)(bptr1 + ke + 8);
            b2.h[0] = *(const v8bf*)(bptr2 + ke); b2.h[1] = *(const v8bf*)(bptr2 + ke + 8);
            b3.h[0] = *(const v8bf*)(bptr3 + ke); b3.h[1] = *(const v8bf*)(bptr3 + ke + 8);

            // Each B fragment feeds two WMMAs (two M slices)
            accA0 = __builtin_amdgcn_wmma_f32_16x16x32_bf16(false, a0.v, false, b0.v, (short)0, accA0, false, false);
            accB0 = __builtin_amdgcn_wmma_f32_16x16x32_bf16(false, a1.v, false, b0.v, (short)0, accB0, false, false);
            accA1 = __builtin_amdgcn_wmma_f32_16x16x32_bf16(false, a0.v, false, b1.v, (short)0, accA1, false, false);
            accB1 = __builtin_amdgcn_wmma_f32_16x16x32_bf16(false, a1.v, false, b1.v, (short)0, accB1, false, false);
            accA2 = __builtin_amdgcn_wmma_f32_16x16x32_bf16(false, a0.v, false, b2.v, (short)0, accA2, false, false);
            accB2 = __builtin_amdgcn_wmma_f32_16x16x32_bf16(false, a1.v, false, b2.v, (short)0, accB2, false, false);
            accA3 = __builtin_amdgcn_wmma_f32_16x16x32_bf16(false, a0.v, false, b3.v, (short)0, accA3, false, false);
            accB3 = __builtin_amdgcn_wmma_f32_16x16x32_bf16(false, a1.v, false, b3.v, (short)0, accB3, false, false);
        }

        aptrA += 128; aptrB += 128;
        bptr0 += 128; bptr1 += 128; bptr2 += 128; bptr3 += 128;
    }

    // Fused gate epilogue: bias + activations + cell update, for both slices.
    const int hcol = h0 + lrow;
    const float bi = bias[0 * LSTM_H + hcol];
    const float bf = bias[1 * LSTM_H + hcol];
    const float bg = bias[2 * LSTM_H + hcol];
    const float bo = bias[3 * LSTM_H + hcol];

#pragma unroll
    for (int j = 0; j < 8; ++j) {
        {
            const int r = mA + j + lhi * 8;
            const size_t off = (size_t)r * LSTM_H + hcol;
            const float fm = fmask[r];
            const float iv = sigmoid_f(accA0[j] + bi);
            const float fv = sigmoid_f(accA1[j] + bf) * fm;
            const float gv = tanhf(accA2[j] + bg);
            const float ov = sigmoid_f(accA3[j] + bo);
            c_pre[off]  = fv * c_prev[off] + iv * gv;
            o_gate[off] = ov;
        }
        {
            const int r = mB + j + lhi * 8;
            const size_t off = (size_t)r * LSTM_H + hcol;
            const float fm = fmask[r];
            const float iv = sigmoid_f(accB0[j] + bi);
            const float fv = sigmoid_f(accB1[j] + bf) * fm;
            const float gv = tanhf(accB2[j] + bg);
            const float ov = sigmoid_f(accB3[j] + bo);
            c_pre[off]  = fv * c_prev[off] + iv * gv;
            o_gate[off] = ov;
        }
    }
}

// ---------------------------------------------------------------------------
// Kernel 4: per-row LayerNorm over c_pre, then h_t = o * tanh(c_t).
//   One 256-thread block per row; deterministic LDS tree reduction.
// ---------------------------------------------------------------------------
__global__ __launch_bounds__(256) void lstm_ln(
    const float* __restrict__ c_pre, const float* __restrict__ o_gate,
    const float* __restrict__ gamma, const float* __restrict__ beta,
    float* __restrict__ out)
{
    __shared__ float s_sum[256];
    __shared__ float s_sq[256];
    const int r = blockIdx.x;
    const float* row = c_pre + (size_t)r * LSTM_H;

    float vals[4];
    float lsum = 0.f, lsq = 0.f;
#pragma unroll
    for (int i = 0; i < 4; ++i) {
        float v = row[threadIdx.x + i * 256];
        vals[i] = v;
        lsum += v;
        lsq  += v * v;
    }
    s_sum[threadIdx.x] = lsum;
    s_sq[threadIdx.x]  = lsq;
    __syncthreads();
    for (int s = 128; s > 0; s >>= 1) {
        if (threadIdx.x < s) {
            s_sum[threadIdx.x] += s_sum[threadIdx.x + s];
            s_sq[threadIdx.x]  += s_sq[threadIdx.x + s];
        }
        __syncthreads();
    }
    const float inv_n = 1.0f / (float)LSTM_H;
    const float mu  = s_sum[0] * inv_n;
    const float var = s_sq[0] * inv_n - mu * mu;   // biased variance (torch LN)
    const float rs  = rsqrtf(var + 1e-5f);

    float* hout = out;                                   // h_t first
    float* cout = out + (size_t)LSTM_B * LSTM_H;         // then c_t
#pragma unroll
    for (int i = 0; i < 4; ++i) {
        const int h = threadIdx.x + i * 256;
        const size_t off = (size_t)r * LSTM_H + h;
        const float cn = (vals[i] - mu) * rs * gamma[h] + beta[h];
        cout[off] = cn;
        hout[off] = o_gate[off] * tanhf(cn);
    }
}

// ---------------------------------------------------------------------------
// Launch: workspace layout (112 MB total):
//   [0,   32MB) : A   bf16 [8192 x 2048]
//   [32,  48MB) : Wt  bf16 [4096 x 2048]
//   [48,  80MB) : c_pre f32 [8192 x 1024]
//   [80, 112MB) : o_gate f32 [8192 x 1024]
// ---------------------------------------------------------------------------
extern "C" void kernel_launch(void* const* d_in, const int* in_sizes, int n_in,
                              void* d_out, int out_size, void* d_ws, size_t ws_size,
                              hipStream_t stream) {
    const float* x      = (const float*)d_in[0];
    const float* hprev  = (const float*)d_in[1];
    const float* cprev  = (const float*)d_in[2];
    const float* fmask  = (const float*)d_in[3];
    const float* W      = (const float*)d_in[4];
    const float* b      = (const float*)d_in[5];
    const float* gamma  = (const float*)d_in[6];
    const float* beta   = (const float*)d_in[7];

    char* ws = (char*)d_ws;
    __hip_bfloat16* A   = (__hip_bfloat16*)(ws);
    __hip_bfloat16* Wt  = (__hip_bfloat16*)(ws + (size_t)LSTM_B * LSTM_K * 2);
    float* c_pre  = (float*)(ws + (size_t)LSTM_B * LSTM_K * 2 + (size_t)LSTM_4H * LSTM_K * 2);
    float* o_gate = c_pre + (size_t)LSTM_B * LSTM_H;

    // 1) pack + convert A
    lstm_convA<<<(LSTM_B * LSTM_K) / 256, 256, 0, stream>>>(x, hprev, A);

    // 2) transpose + convert W
    dim3 tb(32, 8);
    dim3 tg(LSTM_4H / 32, LSTM_K / 32);
    lstm_transposeW<<<tg, tb, 0, stream>>>(W, Wt);

    // 3) WMMA GEMM + fused gates (grid: x over h-blocks of 32, y over row-blocks of 128)
    dim3 gb(256);
    dim3 gg(LSTM_H / 32, LSTM_B / 128);
    lstm_gemm_gates<<<gg, gb, 0, stream>>>(A, Wt, b, cprev, fmask, c_pre, o_gate);

    // 4) LayerNorm + output
    lstm_ln<<<LSTM_B, 256, 0, stream>>>(c_pre, o_gate, gamma, beta, (float*)d_out);
}
